// MultiHeadBiasedAttentionADALN_MM_47519518163041
// MI455X (gfx1250) — matrix operand order repaired and checked by reading, not compiled, not run on hardware
//
#include <hip/hip_runtime.h>

// ---------------------------------------------------------------------------
// Types for CDNA5 WMMA (wave32): v16h A/B operands, v8f accumulator.
// ---------------------------------------------------------------------------
typedef _Float16 half8_t  __attribute__((ext_vector_type(8)));
typedef _Float16 half16_t __attribute__((ext_vector_type(16)));
typedef float    f32x8    __attribute__((ext_vector_type(8)));

union H16 { half16_t v; half8_t h[2]; };

__device__ __forceinline__ f32x8 wmma_f16(half16_t a, half16_t b, f32x8 c) {
    // 8 args: (neg_a, A, neg_b, B, c_mod, C, reuse_a, reuse_b)
    return __builtin_amdgcn_wmma_f32_16x16x32_f16(false, a, false, b, (short)0, c,
                                                  false, false);
}

// A tile 16x32 f16 (row-major, lda elems). lane r = M row, g picks K group.
// element e (0..15): K = e + 16g -> two b128 loads of 16 contiguous K.
__device__ __forceinline__ half16_t load_a_tile(const _Float16* __restrict__ A,
                                                int lda, int lane) {
    int r = lane & 15, g = lane >> 4;
    const _Float16* p = A + r * lda + g * 16;
    H16 u;
    u.h[0] = *(const half8_t*)(p);
    u.h[1] = *(const half8_t*)(p + 8);
    return u.v;
}

// B tile 32x16 f16 from N-major storage Bt[n*ldb + k]. lane r = N column.
// element e (0..15): K = e + 16g -> two b128 loads of 16 contiguous K.
__device__ __forceinline__ half16_t load_b_tile(const _Float16* __restrict__ Bt,
                                                int ldb, int lane) {
    int r = lane & 15, g = lane >> 4;
    const _Float16* p = Bt + r * ldb + g * 16;
    H16 u;
    u.h[0] = *(const half8_t*)(p);
    u.h[1] = *(const half8_t*)(p + 8);
    return u.v;
}

// ---------------------------------------------------------------------------
// Generic WMMA GEMM: out[M,N] = A[M,K](f16) @ Bt[N,K](f16,pre-transposed) + bias
// One wave computes a 32x64 tile (2 M-tiles x 4 N-tiles = 8 accumulators).
// Block = 4 waves. Per K-step: 4 A + 8 B b128 loads for 8 WMMAs.
// MODE 0: plain+bias   MODE 1: sigmoid(acc+bias)   MODE 2: (acc+bias)*aux*mask
// ---------------------------------------------------------------------------
template <int MODE>
__global__ void gemm_kernel(const _Float16* __restrict__ A,
                            const _Float16* __restrict__ Bt,
                            const float* __restrict__ bias,
                            const float* __restrict__ aux,
                            const unsigned char* __restrict__ mask,
                            float* __restrict__ out, int M, int N, int K) {
    int lane = threadIdx.x & 31;
    int w    = threadIdx.x >> 5;
    int t    = blockIdx.x * 4 + w;
    int mtiles = M >> 5;                      // 32-row tiles
    int ncols  = N >> 6;                      // 64-col tiles
    if (t >= mtiles * ncols) return;          // wave-uniform guard
    int mt = t % mtiles;
    int nc = t / mtiles;

    f32x8 acc[2][4] = {};
    for (int k0 = 0; k0 < K; k0 += 32) {
        half16_t a[2];
#pragma unroll
        for (int i = 0; i < 2; ++i)
            a[i] = load_a_tile(A + (size_t)(mt * 32 + i * 16) * K + k0, K, lane);
#pragma unroll
        for (int j = 0; j < 4; ++j) {
            const _Float16* Bp = Bt + (size_t)(nc * 64 + j * 16) * K + k0;
            half16_t b = load_b_tile(Bp, K, lane);
#pragma unroll
            for (int i = 0; i < 2; ++i)
                acc[i][j] = wmma_f16(a[i], b, acc[i][j]);
        }
    }

    int r15 = lane & 15, g = lane >> 4;
#pragma unroll
    for (int i = 0; i < 2; ++i) {
#pragma unroll
        for (int j = 0; j < 4; ++j) {
#pragma unroll
            for (int r = 0; r < 8; ++r) {
                int row = mt * 32 + i * 16 + r + 8 * g;
                int col = nc * 64 + j * 16 + r15;
                float v = acc[i][j][r];
                if (bias) v += bias[col];
                if (MODE == 1) v = 1.0f / (1.0f + __expf(-v));
                if (MODE == 2) {
                    v *= aux[(size_t)row * N + col];
                    v *= mask[row] ? 1.0f : 0.0f;
                }
                out[(size_t)row * N + col] = v;
            }
        }
    }
}

// ---------------------------------------------------------------------------
// cond layer-norm: cn = LN(cond)*cond_ln_w (f16) and raw cond copy (f16).
// One row (512 elems) per 256-thread block.
// ---------------------------------------------------------------------------
__global__ void cond_ln_kernel(const float* __restrict__ cond,
                               const float* __restrict__ clw,
                               _Float16* __restrict__ cnf,
                               _Float16* __restrict__ condf) {
    __shared__ float red[2][8];
    int row = blockIdx.x, tid = threadIdx.x;
    float v0 = cond[row * 512 + tid];
    float v1 = cond[row * 512 + 256 + tid];
    float s1 = v0 + v1, s2 = v0 * v0 + v1 * v1;
    for (int off = 1; off < 32; off <<= 1) {
        s1 += __shfl_xor(s1, off, 32);
        s2 += __shfl_xor(s2, off, 32);
    }
    int wv = tid >> 5, lane = tid & 31;
    if (lane == 0) { red[0][wv] = s1; red[1][wv] = s2; }
    __syncthreads();
    s1 = 0.f; s2 = 0.f;
#pragma unroll
    for (int i = 0; i < 8; ++i) { s1 += red[0][i]; s2 += red[1][i]; }
    float mu = s1 * (1.0f / 512.0f);
    float var = s2 * (1.0f / 512.0f) - mu * mu;
    float rstd = rsqrtf(var + 1e-5f);
    condf[row * 512 + tid]       = (_Float16)v0;
    condf[row * 512 + 256 + tid] = (_Float16)v1;
    cnf[row * 512 + tid]         = (_Float16)((v0 - mu) * rstd * clw[tid]);
    cnf[row * 512 + 256 + tid]   = (_Float16)((v1 - mu) * rstd * clw[256 + tid]);
}

// ---------------------------------------------------------------------------
// xa = (LN(x)*gamma + beta) * mask  -> f16. One row (768) per block of 256.
// ---------------------------------------------------------------------------
__global__ void xa_kernel(const float* __restrict__ x,
                          const float* __restrict__ gamma,
                          const float* __restrict__ beta,
                          const unsigned char* __restrict__ mask,
                          _Float16* __restrict__ xaf) {
    __shared__ float red[2][8];
    int row = blockIdx.x, tid = threadIdx.x;
    float v[3];
    float s1 = 0.f, s2 = 0.f;
#pragma unroll
    for (int e = 0; e < 3; ++e) {
        v[e] = x[row * 768 + e * 256 + tid];
        s1 += v[e]; s2 += v[e] * v[e];
    }
    for (int off = 1; off < 32; off <<= 1) {
        s1 += __shfl_xor(s1, off, 32);
        s2 += __shfl_xor(s2, off, 32);
    }
    int wv = tid >> 5, lane = tid & 31;
    if (lane == 0) { red[0][wv] = s1; red[1][wv] = s2; }
    __syncthreads();
    s1 = 0.f; s2 = 0.f;
#pragma unroll
    for (int i = 0; i < 8; ++i) { s1 += red[0][i]; s2 += red[1][i]; }
    float mu = s1 * (1.0f / 768.0f);
    float var = s2 * (1.0f / 768.0f) - mu * mu;
    float rstd = rsqrtf(var + 1e-5f);
    float mf = mask[row] ? 1.0f : 0.0f;
#pragma unroll
    for (int e = 0; e < 3; ++e) {
        int c = e * 256 + tid;
        float xn = (v[e] - mu) * rstd;
        xaf[row * 768 + c] =
            (_Float16)((xn * gamma[row * 768 + c] + beta[row * 768 + c]) * mf);
    }
}

// ---------------------------------------------------------------------------
// Per-head LN (Dh=64) for q/k: one wave per (token, head). -> f16
// ---------------------------------------------------------------------------
__global__ void head_ln_kernel(const float* __restrict__ src,
                               const float* __restrict__ w,
                               const float* __restrict__ bprm,
                               _Float16* __restrict__ dst) {
    int tid = threadIdx.x, lane = tid & 31, wv = tid >> 5;
    int grp = blockIdx.x * 8 + wv;   // token*12+h, 0..6143
    const float* p = src + (size_t)grp * 64;
    float v0 = p[lane], v1 = p[lane + 32];
    float s1 = v0 + v1, s2 = v0 * v0 + v1 * v1;
    for (int off = 1; off < 32; off <<= 1) {
        s1 += __shfl_xor(s1, off, 32);
        s2 += __shfl_xor(s2, off, 32);
    }
    float mu = s1 * (1.0f / 64.0f);
    float var = s2 * (1.0f / 64.0f) - mu * mu;
    float rstd = rsqrtf(var + 1e-5f);
    dst[(size_t)grp * 64 + lane]      = (_Float16)((v0 - mu) * rstd * w[lane] + bprm[lane]);
    dst[(size_t)grp * 64 + lane + 32] = (_Float16)((v1 - mu) * rstd * w[lane + 32] + bprm[lane + 32]);
}

// ---------------------------------------------------------------------------
// vT pack: vT[h*64+d][640] with 64-key zero padding on each side (absorbs
// clipped window edges; those keys carry ~0 attention weight anyway).
// ---------------------------------------------------------------------------
__global__ void vT_kernel(const float* __restrict__ v, _Float16* __restrict__ vT) {
    int idx = blockIdx.x * 256 + threadIdx.x;
    if (idx >= 12 * 64 * 640) return;
    int kp = idx % 640;
    int hd = idx / 640;
    int key = kp - 64;
    float val = (key >= 0 && key < 512) ? v[(size_t)key * 768 + hd] : 0.0f;
    vT[idx] = (_Float16)val;
}

// Weight transpose + f16 convert: Wt[n*K + k] = W[k*N + n]
__global__ void transpose_f16_kernel(const float* __restrict__ W,
                                     _Float16* __restrict__ Wt, int K, int N) {
    int idx = blockIdx.x * 256 + threadIdx.x;
    if (idx >= K * N) return;
    int k = idx % K, n = idx / K;
    Wt[idx] = (_Float16)W[(size_t)k * N + n];
}

// ---------------------------------------------------------------------------
// Fused windowed pair bias: for each windowed (q,k) pair, LN(pair_rep row,256)
// then 12-wide dot with pair_bias_w; invalid/masked -> -1e9. One wave per row.
// Only 25% of pair_rep is touched (67 MB vs 268 MB) -> this is the HBM-bound
// kernel, so it stays fused VALU (no f16 staging round-trip).
// ---------------------------------------------------------------------------
__global__ void pair_bias_kernel(const float* __restrict__ pair_rep,
                                 const unsigned char* __restrict__ pair_mask,
                                 const float* __restrict__ plw,
                                 const float* __restrict__ plb,
                                 const float* __restrict__ pbw,
                                 float* __restrict__ biasOut) {
    __shared__ float wsm[256 * 12];
    __shared__ float lw[256], lb[256];
    __shared__ float partial[8][32][12];
    int tid = threadIdx.x;
    for (int i = tid; i < 256 * 12; i += 256) wsm[i] = pbw[i];
    lw[tid] = plw[tid];
    lb[tid] = plb[tid];
    __syncthreads();

    int wv = tid >> 5, lane = tid & 31;
    int row  = blockIdx.x * 8 + wv;          // 0..65535 = nb*32*128
    int bblk = row >> 12;
    int rem  = row & 4095;
    int q    = rem >> 7;
    int kcol = rem & 127;
    int i    = bblk * 32 + q;                // query token
    int kraw = bblk * 32 - 48 + kcol;
    int j    = kraw < 0 ? 0 : (kraw > 511 ? 511 : kraw);
    bool valid = (kraw >= 0) && (kraw < 512) && (pair_mask[(size_t)i * 512 + j] != 0);

    const float* pr = pair_rep + ((size_t)i * 512 + j) * 256 + lane * 8;
    float4 t0 = *(const float4*)(pr);
    float4 t1 = *(const float4*)(pr + 4);
    float xv[8] = {t0.x, t0.y, t0.z, t0.w, t1.x, t1.y, t1.z, t1.w};

    float s1 = 0.f, s2 = 0.f;
#pragma unroll
    for (int e = 0; e < 8; ++e) { s1 += xv[e]; s2 += xv[e] * xv[e]; }
    for (int off = 1; off < 32; off <<= 1) {
        s1 += __shfl_xor(s1, off, 32);
        s2 += __shfl_xor(s2, off, 32);
    }
    float mu = s1 * (1.0f / 256.0f);
    float var = s2 * (1.0f / 256.0f) - mu * mu;
    float rstd = rsqrtf(var + 1e-5f);

    float acc[12] = {};
#pragma unroll
    for (int e = 0; e < 8; ++e) {
        int kk = lane * 8 + e;
        float xn = (xv[e] - mu) * rstd * lw[kk] + lb[kk];
#pragma unroll
        for (int hh = 0; hh < 12; ++hh) acc[hh] += xn * wsm[kk * 12 + hh];
    }
#pragma unroll
    for (int hh = 0; hh < 12; ++hh) partial[wv][lane][hh] = acc[hh];
    __syncthreads();
    if (lane < 12) {
        float t = 0.f;
#pragma unroll
        for (int l = 0; l < 32; ++l) t += partial[wv][l][lane];
        biasOut[(size_t)row * 12 + lane] = valid ? t : -1.0e9f;
    }
}

// ---------------------------------------------------------------------------
// Windowed attention: one wave per (block, head, 16-row q-tile).
// 16q x 128k x 64d per wave -> scores s[8] (64 VGPRs), no spills.
// scores = q@k^T (WMMA) * 1/8 + bias ; softmax over k (half-wave shfl);
// probs -> LDS -> A-layout ; o = p@v (WMMA vs padded vT). Output f16.
// Grid: (16 blocks, 6), block = 128 threads = 4 waves = (head,mt) pairs.
// ---------------------------------------------------------------------------
__global__ void attn_kernel(const _Float16* __restrict__ qf,
                            const _Float16* __restrict__ kf,
                            const _Float16* __restrict__ vT,
                            const float* __restrict__ bias,
                            _Float16* __restrict__ of) {
    __shared__ _Float16 smem[4][16 * 128];
    int lane = threadIdx.x & 31;
    int w    = threadIdx.x >> 5;
    int b    = blockIdx.x;                 // 0..15
    int idx  = blockIdx.y * 4 + w;         // 0..23
    int h    = idx >> 1;                   // 0..11
    int mt   = idx & 1;                    // q-tile within block
    int r15 = lane & 15, g = lane >> 4;
    int kstart = b * 32 - 48;

    // ---- scores = q @ k^T : M=16 x N=128 (8 tiles), K=64 ----
    f32x8 s[8] = {};
    for (int kk = 0; kk < 64; kk += 32) {
        half16_t a = load_a_tile(
            qf + (size_t)(b * 32 + mt * 16) * 768 + h * 64 + kk, 768, lane);
#pragma unroll
        for (int nt = 0; nt < 8; ++nt) {
            int key = kstart + nt * 16 + r15;
            key = key < 0 ? 0 : (key > 511 ? 511 : key);
            const _Float16* p = kf + (size_t)key * 768 + h * 64 + kk + g * 16;
            H16 u;
            u.h[0] = *(const half8_t*)(p);
            u.h[1] = *(const half8_t*)(p + 8);
            s[nt] = wmma_f16(a, u.v, s[nt]);
        }
    }

    // ---- scale + bias ----
    const float scale = 0.125f;  // 1/sqrt(64)
#pragma unroll
    for (int nt = 0; nt < 8; ++nt)
#pragma unroll
        for (int r = 0; r < 8; ++r) {
            int qq = mt * 16 + r + 8 * g;
            int kc = nt * 16 + r15;
            s[nt][r] = s[nt][r] * scale +
                       bias[((size_t)(b * 32 + qq) * 128 + kc) * 12 + h];
        }

    // ---- softmax over k: each row lives in 16 lanes x 8 accs ----
#pragma unroll
    for (int r = 0; r < 8; ++r) {
        float m = -1.0e30f;
#pragma unroll
        for (int nt = 0; nt < 8; ++nt) m = fmaxf(m, s[nt][r]);
        for (int off = 1; off < 16; off <<= 1) m = fmaxf(m, __shfl_xor(m, off, 32));
        float sum = 0.f;
#pragma unroll
        for (int nt = 0; nt < 8; ++nt) {
            float e = __expf(s[nt][r] - m);
            s[nt][r] = e;
            sum += e;
        }
        for (int off = 1; off < 16; off <<= 1) sum += __shfl_xor(sum, off, 32);
        float inv = 1.0f / sum;
#pragma unroll
        for (int nt = 0; nt < 8; ++nt) s[nt][r] *= inv;
    }

    // ---- probs to LDS [q(16)][k(128)] f16 ----
    _Float16* at = smem[w];
#pragma unroll
    for (int nt = 0; nt < 8; ++nt)
#pragma unroll
        for (int r = 0; r < 8; ++r) {
            int ql = r + 8 * g;            // local q row 0..15
            int kc = nt * 16 + r15;
            at[ql * 128 + kc] = (_Float16)s[nt][r];
        }
    __syncthreads();

    // ---- o = p @ v : M=16 x N=64 (4 tiles), K=128 ----
    f32x8 co[4] = {};
    for (int k0 = 0; k0 < 128; k0 += 32) {
        half16_t a;
        {
            const _Float16* p = at + r15 * 128 + k0 + 16 * g;
            H16 u;
            u.h[0] = *(const half8_t*)(p);
            u.h[1] = *(const half8_t*)(p + 8);
            a = u.v;
        }
#pragma unroll
        for (int nt = 0; nt < 4; ++nt) {
            const _Float16* p =
                vT + (size_t)(h * 64 + nt * 16 + r15) * 640 + 64 + kstart + k0 + 16 * g;
            H16 u;
            u.h[0] = *(const half8_t*)(p);
            u.h[1] = *(const half8_t*)(p + 8);
            co[nt] = wmma_f16(a, u.v, co[nt]);
        }
    }
#pragma unroll
    for (int nt = 0; nt < 4; ++nt)
#pragma unroll
        for (int r = 0; r < 8; ++r) {
            int qq = b * 32 + mt * 16 + r + 8 * g;
            int d  = nt * 16 + r15;
            of[(size_t)qq * 768 + h * 64 + d] = (_Float16)co[nt][r];
        }
}

// ---------------------------------------------------------------------------
// Host launch
// ---------------------------------------------------------------------------
extern "C" void kernel_launch(void* const* d_in, const int* in_sizes, int n_in,
                              void* d_out, int out_size, void* d_ws, size_t ws_size,
                              hipStream_t stream) {
    const float* x    = (const float*)d_in[0];
    const float* prp  = (const float*)d_in[1];
    const float* cond = (const float*)d_in[2];
    const unsigned char* mask  = (const unsigned char*)d_in[3];
    const unsigned char* pmask = (const unsigned char*)d_in[4];
    const float* agw = (const float*)d_in[5];
    const float* agb = (const float*)d_in[6];
    const float* abw = (const float*)d_in[7];
    const float* clw = (const float*)d_in[8];
    const float* wq  = (const float*)d_in[9];
    const float* bq  = (const float*)d_in[10];
    const float* wk  = (const float*)d_in[11];
    const float* bk  = (const float*)d_in[12];
    const float* wv  = (const float*)d_in[13];
    const float* bv  = (const float*)d_in[14];
    const float* qlw = (const float*)d_in[15];
    const float* qlb = (const float*)d_in[16];
    const float* klw = (const float*)d_in[17];
    const float* klb = (const float*)d_in[18];
    const float* plw = (const float*)d_in[19];
    const float* plb = (const float*)d_in[20];
    const float* pbw = (const float*)d_in[21];
    const float* wo  = (const float*)d_in[22];
    const float* bo  = (const float*)d_in[23];
    const float* sgw = (const float*)d_in[24];
    const float* sgb = (const float*)d_in[25];
    (void)in_sizes; (void)n_in; (void)out_size; (void)ws_size;

    char* ws = (char*)d_ws;
    size_t off = 0;
    auto alloc = [&](size_t bytes) -> void* {
        void* p = ws + off;
        off += (bytes + 255) & ~(size_t)255;
        return p;
    };

    _Float16* cnf   = (_Float16*)alloc(512 * 512 * 2);
    _Float16* condf = (_Float16*)alloc(512 * 512 * 2);
    _Float16* agwT  = (_Float16*)alloc(768 * 512 * 2);
    _Float16* abwT  = (_Float16*)alloc(768 * 512 * 2);
    _Float16* sgwT  = (_Float16*)alloc(768 * 512 * 2);
    _Float16* wqT   = (_Float16*)alloc(768 * 768 * 2);
    _Float16* wkT   = (_Float16*)alloc(768 * 768 * 2);
    _Float16* wvT   = (_Float16*)alloc(768 * 768 * 2);
    _Float16* woT   = (_Float16*)alloc(768 * 768 * 2);
    float* gamma    = (float*)alloc(512 * 768 * 4);
    float* beta     = (float*)alloc(512 * 768 * 4);
    float* gbuf     = (float*)alloc(512 * 768 * 4);
    _Float16* xaf   = (_Float16*)alloc(512 * 768 * 2);
    float* qf32     = (float*)alloc(512 * 768 * 4);
    float* kf32     = (float*)alloc(512 * 768 * 4);
    float* vf32     = (float*)alloc(512 * 768 * 4);
    _Float16* qf16  = (_Float16*)alloc(512 * 768 * 2);
    _Float16* kf16  = (_Float16*)alloc(512 * 768 * 2);
    _Float16* vTb   = (_Float16*)alloc(12 * 64 * 640 * 2);
    float* biasb    = (float*)alloc((size_t)65536 * 12 * 4);
    _Float16* of16  = (_Float16*)alloc(512 * 768 * 2);

    // Weight transposes (f32 -> f16, N-major for WMMA B operand)
    transpose_f16_kernel<<<(768 * 512 + 255) / 256, 256, 0, stream>>>(agw, agwT, 512, 768);
    transpose_f16_kernel<<<(768 * 512 + 255) / 256, 256, 0, stream>>>(abw, abwT, 512, 768);
    transpose_f16_kernel<<<(768 * 512 + 255) / 256, 256, 0, stream>>>(sgw, sgwT, 512, 768);
    transpose_f16_kernel<<<(768 * 768 + 255) / 256, 256, 0, stream>>>(wq, wqT, 768, 768);
    transpose_f16_kernel<<<(768 * 768 + 255) / 256, 256, 0, stream>>>(wk, wkT, 768, 768);
    transpose_f16_kernel<<<(768 * 768 + 255) / 256, 256, 0, stream>>>(wv, wvT, 768, 768);
    transpose_f16_kernel<<<(768 * 768 + 255) / 256, 256, 0, stream>>>(wo, woT, 768, 768);

    // cond LN
    cond_ln_kernel<<<512, 256, 0, stream>>>(cond, clw, cnf, condf);

    // adaLN gamma (sigmoid), beta, scale gate g (sigmoid): M=512 N=768 K=512
    // tiles = (512/32)*(768/64) = 16*12 = 192 waves -> 48 blocks
    gemm_kernel<1><<<48, 128, 0, stream>>>(cnf, agwT, agb, nullptr, nullptr, gamma, 512, 768, 512);
    gemm_kernel<0><<<48, 128, 0, stream>>>(cnf, abwT, nullptr, nullptr, nullptr, beta, 512, 768, 512);
    gemm_kernel<1><<<48, 128, 0, stream>>>(condf, sgwT, sgb, nullptr, nullptr, gbuf, 512, 768, 512);

    // xa = (LN(x)*gamma + beta) * mask
    xa_kernel<<<512, 256, 0, stream>>>(x, gamma, beta, mask, xaf);

    // QKV projections
    gemm_kernel<0><<<48, 128, 0, stream>>>(xaf, wqT, bq, nullptr, nullptr, qf32, 512, 768, 768);
    gemm_kernel<0><<<48, 128, 0, stream>>>(xaf, wkT, bk, nullptr, nullptr, kf32, 512, 768, 768);
    gemm_kernel<0><<<48, 128, 0, stream>>>(xaf, wvT, bv, nullptr, nullptr, vf32, 512, 768, 768);

    // per-head LN for q,k ; vT pack (padded)
    head_ln_kernel<<<768, 256, 0, stream>>>(qf32, qlw, qlb, qf16);
    head_ln_kernel<<<768, 256, 0, stream>>>(kf32, klw, klb, kf16);
    vT_kernel<<<(12 * 64 * 640 + 255) / 256, 256, 0, stream>>>(vf32, vTb);

    // windowed pair bias (fused LN + 12-dot), masked -> -1e9
    pair_bias_kernel<<<8192, 256, 0, stream>>>(prp, pmask, plw, plb, pbw, biasb);

    // attention: 16 blocks x 12 heads x 2 q-tiles (4 waves per 128-thr WG)
    attn_kernel<<<dim3(16, 6), 128, 0, stream>>>(qf16, kf16, vTb, biasb, of16);

    // out = (o @ wo + bo) * g * mask
    gemm_kernel<2><<<48, 128, 0, stream>>>(of16, woT, bo, gbuf, mask, (float*)d_out,
                                           512, 768, 768);
}
